// SingleTaskNet_78099685311017
// MI455X (gfx1250) — matrix operand order repaired and checked by reading, not compiled, not run on hardware
//
#include <hip/hip_runtime.h>
#include <hip/hip_bf16.h>
#include <math.h>

typedef _Float16 half_t;
typedef __attribute__((ext_vector_type(16))) _Float16 v16h;
typedef __attribute__((ext_vector_type(8)))  _Float16 v8h;
typedef __attribute__((ext_vector_type(8)))  float    v8f;

#define N_NODES 50000
#define N_EDGES 800000
#define N_GRAPHS 512
#define FEA 128
#define NODE_IN 92
#define EDGE_IN 41
#define BN_EPS 1e-5f

#define CG_WAVES 16                       // 512-thread workgroup for the edge kernel
#define CG_LDS_BYTES (2 * 12 * 4096 * 2)  // Wf + Ws packed f16 = 196608 B

// ---------------------------------------------------------------- device utils

__device__ __forceinline__ v8f wmma_f16(v16h a, v16h b, v8f c) {
  // D = A(16x32 f16) x B(32x16 f16) + C(16x16 f32)
  return __builtin_amdgcn_wmma_f32_16x16x32_f16(false, a, false, b, (short)0, c,
                                                false, false);
}

// A-fragment (16x32 f16) from a row-major f16 row of >=k0+32 cols.
// ISA layout: lane L (r=L&15,g=L>>4) elem e<8 -> K=k0+8g+e ; e>=8 -> K=k0+8+8g+e
__device__ __forceinline__ v16h load_a_f16(const half_t* __restrict__ row, int k0, int g) {
  const v8h* p0 = (const v8h*)(row + k0 + 8 * g);
  const v8h* p1 = (const v8h*)(row + k0 + 16 + 8 * g);
  v8h lo = *p0, hi = *p1;
  v16h a;
#pragma unroll
  for (int i = 0; i < 8; ++i) { a[i] = lo[i]; a[i + 8] = hi[i]; }
  return a;
}

// B-fragment from fragment-order packed weights: [ks][nt][lane][16 halves]
template <typename P>
__device__ __forceinline__ v16h load_b_packed(const P* packed, int ks, int nt, int lane) {
  const P* base = packed + ((((size_t)ks * 8 + nt) * 32 + lane) << 4);
  v16h b;
#pragma unroll
  for (int i = 0; i < 8; ++i) { b[i] = base[i]; b[i + 8] = base[i + 8]; }
  return b;
}

__device__ __forceinline__ float leaky_f(float x) { return x > 0.f ? x : 0.01f * x; }
__device__ __forceinline__ float sigmoid_f(float x) { return 1.f / (1.f + __expf(-x)); }
__device__ __forceinline__ float softplus_f(float x) { return x > 20.f ? x : log1pf(__expf(x)); }

// ---------------------------------------------------------------- small kernels

__global__ void zero_f32_kernel(float* __restrict__ p, size_t n) {
  size_t i = (size_t)blockIdx.x * blockDim.x + threadIdx.x;
  if (i < n) p[i] = 0.f;
}

// Pack W[K,128] (f32, row-major) into f16 WMMA B-fragment order with K padded.
__global__ void pack_weight_kernel(const float* __restrict__ W, half_t* __restrict__ out,
                                   int K, int total) {
  int idx = blockIdx.x * blockDim.x + threadIdx.x;
  if (idx >= total) return;
  int e    = idx & 15;
  int lane = (idx >> 4) & 31;
  int nt   = (idx >> 9) & 7;
  int ks   = idx >> 12;
  int k = ks * 32 + (e < 8 ? 8 * (lane >> 4) + e : 16 + 8 * (lane >> 4) + (e - 8));   // B layout: K = 8*g + e (e<8), 16 + 8*g + e-8 (e>=8) within 32-chunk
  int n = nt * 16 + (lane & 15);
  out[idx] = (k < K) ? (half_t)W[k * 128 + n] : (half_t)0.f;
}

__global__ void count_kernel(const int* __restrict__ dst, float* __restrict__ cnt) {
  int e = blockIdx.x * blockDim.x + threadIdx.x;
  if (e < N_EDGES) unsafeAtomicAdd(&cnt[dst[e]], 1.f);
}

// ---------------------------------------------------------------- embed GEMMs

// h = leaky(x[N,92] @ W + b) -> h f32 + h f16
__global__ void embed_node_kernel(const float* __restrict__ x, const half_t* __restrict__ Wp,
                                  const float* __restrict__ bias, float* __restrict__ hout,
                                  half_t* __restrict__ hf16) {
  int lane = threadIdx.x & 31, g = lane >> 4, r = lane & 15;
  int tile = blockIdx.x * (blockDim.x >> 5) + (threadIdx.x >> 5);
  if (tile >= N_NODES / 16) return;
  const float* row = x + (size_t)(tile * 16 + r) * NODE_IN;
  v8f C[8];
#pragma unroll
  for (int nt = 0; nt < 8; ++nt)
#pragma unroll
    for (int m = 0; m < 8; ++m) C[nt][m] = 0.f;
  for (int ks = 0; ks < 3; ++ks) {            // K padded 92 -> 96
    v16h a;
#pragma unroll
    for (int e = 0; e < 16; ++e) {
      int k = ks * 32 + 8 * g + e + (e < 8 ? 0 : 8);
      a[e] = (k < NODE_IN) ? (half_t)row[k] : (half_t)0.f;
    }
#pragma unroll
    for (int nt = 0; nt < 8; ++nt)
      C[nt] = wmma_f16(a, load_b_packed(Wp, ks, nt, lane), C[nt]);
  }
#pragma unroll
  for (int nt = 0; nt < 8; ++nt) {
    int col = nt * 16 + r;
    float bv = bias[col];
#pragma unroll
    for (int m = 0; m < 8; ++m) {
      int rr = tile * 16 + 8 * g + m;
      float v = leaky_f(C[nt][m] + bv);
      hout[(size_t)rr * FEA + col] = v;
      hf16[(size_t)rr * FEA + col] = (half_t)v;
    }
  }
}

// e = leaky(edge_attr[E,41] @ W + b) -> f16 only
__global__ void embed_edge_kernel(const float* __restrict__ ea, const half_t* __restrict__ Wp,
                                  const float* __restrict__ bias, half_t* __restrict__ ef16) {
  int lane = threadIdx.x & 31, g = lane >> 4, r = lane & 15;
  int tile = blockIdx.x * (blockDim.x >> 5) + (threadIdx.x >> 5);
  if (tile >= N_EDGES / 16) return;
  const float* row = ea + (size_t)(tile * 16 + r) * EDGE_IN;
  v8f C[8];
#pragma unroll
  for (int nt = 0; nt < 8; ++nt)
#pragma unroll
    for (int m = 0; m < 8; ++m) C[nt][m] = 0.f;
  for (int ks = 0; ks < 2; ++ks) {            // K padded 41 -> 64
    v16h a;
#pragma unroll
    for (int e = 0; e < 16; ++e) {
      int k = ks * 32 + 8 * g + e + (e < 8 ? 0 : 8);
      a[e] = (k < EDGE_IN) ? (half_t)row[k] : (half_t)0.f;
    }
#pragma unroll
    for (int nt = 0; nt < 8; ++nt)
      C[nt] = wmma_f16(a, load_b_packed(Wp, ks, nt, lane), C[nt]);
  }
#pragma unroll
  for (int nt = 0; nt < 8; ++nt) {
    int col = nt * 16 + r;
    float bv = bias[col];
#pragma unroll
    for (int m = 0; m < 8; ++m) {
      int rr = tile * 16 + 8 * g + m;
      ef16[(size_t)rr * FEA + col] = (half_t)leaky_f(C[nt][m] + bv);
    }
  }
}

// ---------------------------------------------------------------- CGConv edge stage
// z = [h[dst] | h[src] | e] (K=384), msg = sigmoid(zWf+bf)*softplus(zWs+bs),
// scatter-add into sum[dst]. 512-thread block stages Wf+Ws (192KB f16) into LDS;
// each of 16 waves owns one 16-edge tile with all 12 A-fragments prefetched.
__global__ void cgconv_edge_kernel(const half_t* __restrict__ hf16,
                                   const half_t* __restrict__ ef16,
                                   const int* __restrict__ src, const int* __restrict__ dst,
                                   const half_t* __restrict__ Wfp, const half_t* __restrict__ Wsp,
                                   const float* __restrict__ bf, const float* __restrict__ bs,
                                   float* __restrict__ sum) {
  extern __shared__ char smem_raw[];
  half_t* sWf = (half_t*)smem_raw;          // 12*4096 halves (98304 B)
  half_t* sWs = sWf + 12 * 4096;

  // cooperative stage of both weight matrices into LDS (before any divergence)
  {
    int4* dWf = (int4*)sWf;
    int4* dWs = (int4*)sWs;
    const int4* gWf = (const int4*)Wfp;
    const int4* gWs = (const int4*)Wsp;
    for (int i = threadIdx.x; i < 6144; i += blockDim.x) dWf[i] = gWf[i];
    for (int i = threadIdx.x; i < 6144; i += blockDim.x) dWs[i] = gWs[i];
  }
  __syncthreads();

  int lane = threadIdx.x & 31, g = lane >> 4, r = lane & 15;
  int tile = blockIdx.x * CG_WAVES + (threadIdx.x >> 5);
  if (tile >= N_EDGES / 16) return;
  int eA = tile * 16 + r;
  const half_t* rowD = hf16 + (size_t)dst[eA] * FEA;
  const half_t* rowS = hf16 + (size_t)src[eA] * FEA;
  const half_t* rowE = ef16 + (size_t)eA * FEA;

  // prefetch all 12 A fragments (24 b128 gathers issued up front)
  v16h a[12];
#pragma unroll
  for (int ks = 0; ks < 12; ++ks) {
    const half_t* rp = (ks < 4) ? rowD : (ks < 8 ? rowS : rowE);
    a[ks] = load_a_f16(rp, (ks & 3) * 32, g);
  }

  v8f Cf[8], Cs[8];
#pragma unroll
  for (int nt = 0; nt < 8; ++nt)
#pragma unroll
    for (int m = 0; m < 8; ++m) { Cf[nt][m] = 0.f; Cs[nt][m] = 0.f; }

#pragma unroll
  for (int ks = 0; ks < 12; ++ks) {
#pragma unroll
    for (int nt = 0; nt < 8; ++nt) {
      Cf[nt] = wmma_f16(a[ks], load_b_packed(sWf, ks, nt, lane), Cf[nt]);
      Cs[nt] = wmma_f16(a[ks], load_b_packed(sWs, ks, nt, lane), Cs[nt]);
    }
  }

  int dn[8];
#pragma unroll
  for (int m = 0; m < 8; ++m) dn[m] = dst[tile * 16 + 8 * g + m];
#pragma unroll
  for (int nt = 0; nt < 8; ++nt) {
    int col = nt * 16 + r;
    float bfv = bf[col], bsv = bs[col];
#pragma unroll
    for (int m = 0; m < 8; ++m) {
      float msg = sigmoid_f(Cf[nt][m] + bfv) * softplus_f(Cs[nt][m] + bsv);
      unsafeAtomicAdd(sum + (size_t)dn[m] * FEA + col, msg);
    }
  }
}

// aggr = sum/max(cnt,1); BN; h += ; refresh f16 copy
__global__ void node_update_kernel(const float* __restrict__ sum, const float* __restrict__ cnt,
                                   const float* __restrict__ bg, const float* __restrict__ bb,
                                   const float* __restrict__ bm, const float* __restrict__ bv,
                                   float* __restrict__ h, half_t* __restrict__ hf16) {
  int i = blockIdx.x * blockDim.x + threadIdx.x;
  if (i >= N_NODES * FEA) return;
  int n = i >> 7, c = i & 127;
  float aggr = sum[i] / fmaxf(cnt[n], 1.f);
  float bn = (aggr - bm[c]) * rsqrtf(bv[c] + BN_EPS) * bg[c] + bb[c];
  float v = bn + h[i];
  h[i] = v;
  hf16[i] = (half_t)v;
}

// ---------------------------------------------------------------- generic 128x128 GEMM
// out = A_f16[M,128] @ Wp + bias. mode 0: f32 out, mode 1: f16 out. No activation.
__global__ void gemm128_kernel(const half_t* __restrict__ A, const half_t* __restrict__ Wp,
                               const float* __restrict__ bias, float* __restrict__ outf,
                               half_t* __restrict__ outh, int Mtiles, int mode) {
  int lane = threadIdx.x & 31, g = lane >> 4, r = lane & 15;
  int tile = blockIdx.x * (blockDim.x >> 5) + (threadIdx.x >> 5);
  if (tile >= Mtiles) return;
  const half_t* row = A + (size_t)(tile * 16 + r) * FEA;
  v8f C[8];
#pragma unroll
  for (int nt = 0; nt < 8; ++nt)
#pragma unroll
    for (int m = 0; m < 8; ++m) C[nt][m] = 0.f;
#pragma unroll
  for (int ks = 0; ks < 4; ++ks) {
    v16h a = load_a_f16(row, ks * 32, g);
#pragma unroll
    for (int nt = 0; nt < 8; ++nt)
      C[nt] = wmma_f16(a, load_b_packed(Wp, ks, nt, lane), C[nt]);
  }
#pragma unroll
  for (int nt = 0; nt < 8; ++nt) {
    int col = nt * 16 + r;
    float bv = bias[col];
#pragma unroll
    for (int m = 0; m < 8; ++m) {
      int rr = tile * 16 + 8 * g + m;
      float v = C[nt][m] + bv;
      if (mode == 0) outf[(size_t)rr * FEA + col] = v;
      else           outh[(size_t)rr * FEA + col] = (half_t)v;
    }
  }
}

// gate = gate1[N,128] @ W2[128,1] + b2
__global__ void gate2_kernel(const float* __restrict__ g1, const float* __restrict__ W2,
                             const float* __restrict__ b2, float* __restrict__ gate) {
  int n = blockIdx.x * blockDim.x + threadIdx.x;
  if (n >= N_NODES) return;
  float acc = b2[0];
  const float* row = g1 + (size_t)n * FEA;
#pragma unroll 4
  for (int c = 0; c < FEA; ++c) acc += row[c] * W2[c];
  gate[n] = acc;
}

// ---------------------------------------------------------------- pooling
__global__ void pool_max_kernel(const float* __restrict__ gate, const int* __restrict__ batch,
                                unsigned int* __restrict__ gmax_bits) {
  int n = blockIdx.x * blockDim.x + threadIdx.x;
  if (n >= N_NODES) return;
  unsigned int b = __float_as_uint(gate[n]);
  unsigned int key = (b & 0x80000000u) ? ~b : (b | 0x80000000u);
  atomicMax(&gmax_bits[batch[n]], key);
}

__global__ void pool_exp_kernel(const float* __restrict__ gate, const int* __restrict__ batch,
                                const unsigned int* __restrict__ gmax_bits,
                                float* __restrict__ ex, float* __restrict__ denom) {
  int n = blockIdx.x * blockDim.x + threadIdx.x;
  if (n >= N_NODES) return;
  unsigned int key = gmax_bits[batch[n]];
  unsigned int b = (key & 0x80000000u) ? (key & 0x7fffffffu) : ~key;
  float e = __expf(gate[n] - __uint_as_float(b));
  ex[n] = e;
  unsafeAtomicAdd(&denom[batch[n]], e);
}

__global__ void pool_sum_kernel(const float* __restrict__ ex, const float* __restrict__ denom,
                                const float* __restrict__ feat, const int* __restrict__ batch,
                                float* __restrict__ gpool) {
  size_t i = (size_t)blockIdx.x * blockDim.x + threadIdx.x;
  if (i >= (size_t)N_NODES * FEA) return;
  int n = (int)(i >> 7), c = (int)(i & 127);
  int gr = batch[n];
  float alpha = ex[n] / fmaxf(denom[gr], 1e-16f);
  unsafeAtomicAdd(&gpool[(size_t)gr * FEA + c], alpha * feat[i]);
}

// ---------------------------------------------------------------- head MLP (tiny)
__global__ void head_kernel(const float* __restrict__ gpool,
                            const float* __restrict__ W1, const float* __restrict__ b1,
                            const float* __restrict__ W2, const float* __restrict__ b2,
                            const float* __restrict__ Wu, const float* __restrict__ bu,
                            float* __restrict__ out) {
  __shared__ float sg[FEA];
  __shared__ float st[FEA];
  int t = threadIdx.x, gr = blockIdx.x;
  sg[t] = gpool[(size_t)gr * FEA + t];
  __syncthreads();
  float acc = b1[t];
  for (int c = 0; c < FEA; ++c) acc += sg[c] * W1[c * FEA + t];
  st[t] = leaky_f(acc);
  __syncthreads();
  acc = b2[t];
  for (int c = 0; c < FEA; ++c) acc += st[c] * W2[c * FEA + t];
  sg[t] = leaky_f(acc);
  __syncthreads();
  if (t < 4) {
    float o = bu[t];
    for (int c = 0; c < FEA; ++c) o += sg[c] * Wu[c * 4 + t];
    float res = (t == 0) ? o : (t == 2 ? softplus_f(o) + 1.f : softplus_f(o));
    out[gr * 4 + t] = res;
  }
}

// ---------------------------------------------------------------- host launch

extern "C" void kernel_launch(void* const* d_in, const int* in_sizes, int n_in,
                              void* d_out, int out_size, void* d_ws, size_t ws_size,
                              hipStream_t stream) {
  // Inputs in setup_inputs() insertion order, params flattened in insertion order.
  const float* x         = (const float*)d_in[0];
  const float* edge_attr = (const float*)d_in[1];
  const int*   src       = (const int*)d_in[2];              // edge_index[0]
  const int*   dst       = ((const int*)d_in[2]) + N_EDGES;  // edge_index[1]
  const int*   batch     = (const int*)d_in[3];
  const float* node_W = (const float*)d_in[4];
  const float* node_b = (const float*)d_in[5];
  const float* edge_W = (const float*)d_in[6];
  const float* edge_b = (const float*)d_in[7];
  const float *Wf[3], *bf[3], *Ws[3], *bs[3], *bg[3], *bb[3], *bm[3], *bv[3];
  for (int i = 0; i < 3; ++i) {
    int base = 8 + 8 * i;
    Wf[i] = (const float*)d_in[base + 0]; bf[i] = (const float*)d_in[base + 1];
    Ws[i] = (const float*)d_in[base + 2]; bs[i] = (const float*)d_in[base + 3];
    bg[i] = (const float*)d_in[base + 4]; bb[i] = (const float*)d_in[base + 5];
    bm[i] = (const float*)d_in[base + 6]; bv[i] = (const float*)d_in[base + 7];
  }
  const float* gate_W1 = (const float*)d_in[32];
  const float* gate_b1 = (const float*)d_in[33];
  const float* gate_W2 = (const float*)d_in[34];
  const float* gate_b2 = (const float*)d_in[35];
  const float* nn_W1 = (const float*)d_in[36];
  const float* nn_b1 = (const float*)d_in[37];
  const float* nn_W2 = (const float*)d_in[38];
  const float* nn_b2 = (const float*)d_in[39];
  const float* hs_W1 = (const float*)d_in[40];
  const float* hs_b1 = (const float*)d_in[41];
  const float* hs_W2 = (const float*)d_in[42];
  const float* hs_b2 = (const float*)d_in[43];
  const float* unc_W = (const float*)d_in[44];
  const float* unc_b = (const float*)d_in[45];
  float* out = (float*)d_out;

  // ---- workspace carve-out
  char* ws = (char*)d_ws;
  size_t cur = 0;
  auto alloc = [&](size_t bytes) -> char* {
    cur = (cur + 255) & ~(size_t)255;
    char* p = ws + cur;
    cur += bytes;
    return p;
  };
  float*  h      = (float*)alloc((size_t)N_NODES * FEA * 4);
  half_t* hf16   = (half_t*)alloc((size_t)N_NODES * FEA * 2);
  half_t* ef16   = (half_t*)alloc((size_t)N_EDGES * FEA * 2);
  float*  sum    = (float*)alloc((size_t)N_NODES * FEA * 4);   // reused as gate1 later
  float*  cnt    = (float*)alloc((size_t)N_NODES * 4);
  half_t* pk_node  = (half_t*)alloc((size_t)3 * 4096 * 2);
  half_t* pk_edge  = (half_t*)alloc((size_t)2 * 4096 * 2);
  half_t* pk_Wf[3], *pk_Ws[3];
  for (int i = 0; i < 3; ++i) {
    pk_Wf[i] = (half_t*)alloc((size_t)12 * 4096 * 2);
    pk_Ws[i] = (half_t*)alloc((size_t)12 * 4096 * 2);
  }
  half_t* pk_g1  = (half_t*)alloc((size_t)4 * 4096 * 2);
  half_t* pk_n1  = (half_t*)alloc((size_t)4 * 4096 * 2);
  half_t* pk_n2  = (half_t*)alloc((size_t)4 * 4096 * 2);
  float*  gate   = (float*)alloc((size_t)N_NODES * 4);
  float*  ex     = (float*)alloc((size_t)N_NODES * 4);
  half_t* feat1h = (half_t*)alloc((size_t)N_NODES * FEA * 2);
  float*  feat   = (float*)alloc((size_t)N_NODES * FEA * 4);
  unsigned int* gmax = (unsigned int*)alloc((size_t)N_GRAPHS * 4);
  float*  denom  = (float*)alloc((size_t)N_GRAPHS * 4);
  float*  gpool  = (float*)alloc((size_t)N_GRAPHS * FEA * 4);
  (void)ws_size; (void)n_in; (void)in_sizes; (void)out_size;

  const int NF = N_NODES * FEA;              // 6.4M
  const int nfBlocks = (NF + 255) / 256;
  const int nodeTiles = N_NODES / 16, edgeTiles = N_EDGES / 16;
  const int nodeGemmBlocks = (nodeTiles + 3) / 4;   // 4 waves/block
  const int edgeGemmBlocks = (edgeTiles + 3) / 4;
  const int cgBlocks = (edgeTiles + CG_WAVES - 1) / CG_WAVES;   // 16 waves/block

  // allow 192KB dynamic LDS for the edge kernel (idempotent host-side attr,
  // not a stream op -> graph-capture safe; no static guards per harness rules)
  hipFuncSetAttribute((const void*)cgconv_edge_kernel,
                      hipFuncAttributeMaxDynamicSharedMemorySize, CG_LDS_BYTES);

  // ---- pack all weights into WMMA B-fragment order (f16)
  struct PackJob { const float* W; half_t* out; int K; int total; };
  PackJob jobs[] = {
    { node_W, pk_node, NODE_IN, 3 * 4096 },
    { edge_W, pk_edge, EDGE_IN, 2 * 4096 },
    { Wf[0], pk_Wf[0], 384, 12 * 4096 }, { Ws[0], pk_Ws[0], 384, 12 * 4096 },
    { Wf[1], pk_Wf[1], 384, 12 * 4096 }, { Ws[1], pk_Ws[1], 384, 12 * 4096 },
    { Wf[2], pk_Wf[2], 384, 12 * 4096 }, { Ws[2], pk_Ws[2], 384, 12 * 4096 },
    { gate_W1, pk_g1, FEA, 4 * 4096 },
    { nn_W1,  pk_n1, FEA, 4 * 4096 },
    { nn_W2,  pk_n2, FEA, 4 * 4096 },
  };
  for (auto& j : jobs)
    pack_weight_kernel<<<(j.total + 255) / 256, 256, 0, stream>>>(j.W, j.out, j.K, j.total);

  // ---- zero accumulators (cnt, pooling buffers)
  zero_f32_kernel<<<(N_NODES + 255) / 256, 256, 0, stream>>>(cnt, N_NODES);
  zero_f32_kernel<<<1, 256, 0, stream>>>((float*)gmax, N_GRAPHS);   // key 0 == -NaN sentinel
  zero_f32_kernel<<<1, 256, 0, stream>>>(denom, N_GRAPHS);
  zero_f32_kernel<<<(N_GRAPHS * FEA + 255) / 256, 256, 0, stream>>>(gpool, (size_t)N_GRAPHS * FEA);

  // ---- embeds + per-dst edge counts
  embed_node_kernel<<<nodeGemmBlocks, 128, 0, stream>>>(x, pk_node, node_b, h, hf16);
  embed_edge_kernel<<<edgeGemmBlocks, 128, 0, stream>>>(edge_attr, pk_edge, edge_b, ef16);
  count_kernel<<<(N_EDGES + 255) / 256, 256, 0, stream>>>(dst, cnt);

  // ---- 3 CGConv layers
  for (int l = 0; l < 3; ++l) {
    zero_f32_kernel<<<nfBlocks, 256, 0, stream>>>(sum, (size_t)NF);
    cgconv_edge_kernel<<<cgBlocks, 32 * CG_WAVES, CG_LDS_BYTES, stream>>>(
        hf16, ef16, src, dst, pk_Wf[l], pk_Ws[l], bf[l], bs[l], sum);
    node_update_kernel<<<nfBlocks, 256, 0, stream>>>(sum, cnt, bg[l], bb[l], bm[l], bv[l],
                                                     h, hf16);
  }

  // ---- attention pooling
  float* gate1 = sum;  // reuse
  gemm128_kernel<<<nodeGemmBlocks, 128, 0, stream>>>(hf16, pk_g1, gate_b1, gate1, nullptr,
                                                     nodeTiles, 0);
  gate2_kernel<<<(N_NODES + 255) / 256, 256, 0, stream>>>(gate1, gate_W2, gate_b2, gate);
  gemm128_kernel<<<nodeGemmBlocks, 128, 0, stream>>>(hf16, pk_n1, nn_b1, nullptr, feat1h,
                                                     nodeTiles, 1);
  gemm128_kernel<<<nodeGemmBlocks, 128, 0, stream>>>(feat1h, pk_n2, nn_b2, feat, nullptr,
                                                     nodeTiles, 0);
  pool_max_kernel<<<(N_NODES + 255) / 256, 256, 0, stream>>>(gate, batch, gmax);
  pool_exp_kernel<<<(N_NODES + 255) / 256, 256, 0, stream>>>(gate, batch, gmax, ex, denom);
  pool_sum_kernel<<<nfBlocks, 256, 0, stream>>>(ex, denom, feat, batch, gpool);

  // ---- head MLP + evidential transform
  head_kernel<<<N_GRAPHS, FEA, 0, stream>>>(gpool, hs_W1, hs_b1, hs_W2, hs_b2, unc_W, unc_b, out);
}
